// SlidingTileAttention3D_70265664962745
// MI455X (gfx1250) — compile-verified, matrix operand order checked by reading, not run on hardware
//
#include <hip/hip_runtime.h>

#define DEV __device__ __forceinline__

typedef __attribute__((ext_vector_type(16))) _Float16 v16h;
typedef __attribute__((ext_vector_type(8)))  float    v8f;

// ---- static config (matches reference) ----
// B=1, HID=1024, NH=16, D=64, S=8192
// tiles: TILE=128 tokens, NTILE=64, window WK=12 kv tiles
// seq decomposition: s = ntt*2048 + tt*1024 + nth*256 + th*32 + ntw*8 + tw
//   tile n = ntt*16 + nth*4 + ntw ; in-tile p = tt*64 + th*8 + tw

union Frag {
  v16h h;
  int4 q[2];
  _Float16 e[16];
};
union H8 { int4 v; _Float16 e[8]; };

DEV v8f wmma16(const Frag& a, const Frag& b, v8f c) {
  return __builtin_amdgcn_wmma_f32_16x16x32_f16(
      /*neg_a=*/false, a.h, /*neg_b=*/false, b.h,
      /*c_mod=*/(short)0, c, /*reuse_a=*/false, /*reuse_b=*/false);
}

// Load a 16x32 f16 fragment (A layout; B via [N][K]-transposed storage) from a
// row-major [rows][stride] f16 array. Per CDNA5 ISA 16-bit A layout:
//   lanes 0-15 : M=lane,    K = {0..7, 16..23}
//   lanes 16-31: M=lane-16, K = {8..15, 24..31}
// Both 8-half chunks are contiguous -> two b128 loads.
DEV Frag load_frag(const _Float16* __restrict__ base, int stride, int row, int kc32) {
  const int lane = threadIdx.x & 31;
  const int r  = row + (lane & 15);
  const int k0 = kc32 + ((lane & 16) ? 8 : 0);
  Frag f;
  f.q[0] = *(const int4*)(base + (size_t)r * stride + k0);
  f.q[1] = *(const int4*)(base + (size_t)r * stride + k0 + 16);
  return f;
}

DEV void cvt4(H8& h, int o, float4 f) {
  h.e[o]   = (_Float16)f.x; h.e[o+1] = (_Float16)f.y;
  h.e[o+2] = (_Float16)f.z; h.e[o+3] = (_Float16)f.w;
}

// ---------------------------------------------------------------------------
// Prep 1: elementwise f32 -> f16 (hidden states). 8 elems / thread.
// ---------------------------------------------------------------------------
__global__ __launch_bounds__(256)
void sta3d_cvt_kernel(const float* __restrict__ src, _Float16* __restrict__ dst) {
  const size_t i = ((size_t)blockIdx.x * 256 + threadIdx.x) * 8;
  const float4* s4 = (const float4*)(src + i);
  H8 h;
  cvt4(h, 0, s4[0]); cvt4(h, 4, s4[1]);
  *(int4*)(dst + i) = h.v;
}

// ---------------------------------------------------------------------------
// Prep 2: weight convert + transpose: W[1024(k)][1024(n)] f32 -> WT[n][k] f16.
// 64x64 tile per block through LDS.
// ---------------------------------------------------------------------------
__global__ __launch_bounds__(256)
void sta3d_cvtT_kernel(const float* __restrict__ W, _Float16* __restrict__ WT) {
  __shared__ __align__(16) _Float16 tile[64 * 72];
  const int t  = threadIdx.x;
  const int bk = (blockIdx.x & 15) << 6;   // src row (k) tile
  const int bn = (blockIdx.x >> 4) << 6;   // src col (n) tile
  {
    const int r = t >> 2, c0 = (t & 3) << 4;
    const float4* s4 = (const float4*)(W + (size_t)(bk + r) * 1024 + bn + c0);
    H8 h0, h1;
    cvt4(h0, 0, s4[0]); cvt4(h0, 4, s4[1]);
    cvt4(h1, 0, s4[2]); cvt4(h1, 4, s4[3]);
    *(int4*)&tile[r * 72 + c0]     = h0.v;
    *(int4*)&tile[r * 72 + c0 + 8] = h1.v;
  }
  __syncthreads();
  {
    const int n = t >> 2, k0 = (t & 3) << 4;
    H8 h0, h1;
#pragma unroll
    for (int i = 0; i < 8; ++i) h0.e[i] = tile[(k0 + i) * 72 + n];
#pragma unroll
    for (int i = 0; i < 8; ++i) h1.e[i] = tile[(k0 + 8 + i) * 72 + n];
    _Float16* d = WT + (size_t)(bn + n) * 1024 + bk + k0;
    *(int4*)(d)     = h0.v;
    *(int4*)(d + 8) = h1.v;
  }
}

// ---------------------------------------------------------------------------
// GEMM: C[8192 x 1024] = A16[8192 x 1024] * BT16[1024(n) x 1024(k)]^T
// LDS-free: both operands' fragments are contiguous b128 runs in global (L2).
// Block 256 thr = 8 waves; block tile 128(M) x 256(N); wave tile 64x64.
// MODE 0: scatter f16 into [h][tile][p][d]   (Q, K)
// MODE 1: scatter f16 into [h][tile][d][p]   (V transposed)
// MODE 2: plain fp32 [s][c]                  (final projection)
// ---------------------------------------------------------------------------
template <int MODE>
__global__ __launch_bounds__(256)
void sta3d_gemm_kernel(const _Float16* __restrict__ A, const _Float16* __restrict__ BT,
                       _Float16* __restrict__ dstH, float* __restrict__ dstF) {
  const int wave = threadIdx.x >> 5;
  const int lane = threadIdx.x & 31;
  const int bm = (blockIdx.x & 63) << 7;   // 64 M blocks
  const int bn = (blockIdx.x >> 6) << 8;   // 4  N blocks
  const int wm = bm + (wave >> 2) * 64;
  const int wn = bn + (wave & 3) * 64;

  v8f acc[4][4];
#pragma unroll
  for (int i = 0; i < 4; ++i)
#pragma unroll
    for (int j = 0; j < 4; ++j) acc[i][j] = (v8f){};

  for (int ks = 0; ks < 1024; ks += 32) {
    if (ks + 32 < 1024) {  // hint next K slab toward the WGP (global_prefetch_b8)
      __builtin_prefetch(A  + (size_t)(wm + (lane << 1)) * 1024 + ks + 32, 0, 0);
      __builtin_prefetch(BT + (size_t)(wn + (lane << 1)) * 1024 + ks + 32, 0, 0);
    }
    Frag af[4], bf[4];
#pragma unroll
    for (int i = 0; i < 4; ++i) af[i] = load_frag(A,  1024, wm + i * 16, ks);
#pragma unroll
    for (int j = 0; j < 4; ++j) bf[j] = load_frag(BT, 1024, wn + j * 16, ks);
#pragma unroll
    for (int i = 0; i < 4; ++i)
#pragma unroll
      for (int j = 0; j < 4; ++j) acc[i][j] = wmma16(af[i], bf[j], acc[i][j]);
  }

  // epilogue: C layout -> VGPR r holds (M = base + r + (lane>=16 ? 8:0), N = lane&15)
  const int mo   = (lane & 16) ? 8 : 0;
  const int ncol = lane & 15;
#pragma unroll
  for (int i = 0; i < 4; ++i) {
#pragma unroll
    for (int j = 0; j < 4; ++j) {
      const int col = wn + j * 16 + ncol;   // 0..1023 = h*64 + d
      const int h = col >> 6, d = col & 63;
#pragma unroll
      for (int r = 0; r < 8; ++r) {
        const int srow = wm + i * 16 + mo + r;
        const float val = acc[i][j][r];
        if constexpr (MODE == 2) {
          dstF[(size_t)srow * 1024 + col] = val;
        } else {
          const int ntt = srow >> 11, tt = (srow >> 10) & 1, nth = (srow >> 8) & 3;
          const int th = (srow >> 5) & 7, ntw = (srow >> 3) & 3, tw = srow & 7;
          const int nt = ntt * 16 + nth * 4 + ntw;
          const int p  = tt * 64 + th * 8 + tw;
          if constexpr (MODE == 0)
            dstH[(((size_t)(h * 64 + nt) * 128) + p) * 64 + d] = (_Float16)val;
          else
            dstH[(((size_t)(h * 64 + nt) * 64) + d) * 128 + p] = (_Float16)val;
        }
      }
    }
  }
}

// ---------------------------------------------------------------------------
// Attention: one block per (head, q-tile). 8 waves x 16 query rows.
// Flash-style streaming over the 12 kv tiles of the 3D window.
// Q,K: [h][tile][128][64] f16 ; VT: [h][tile][64][128] f16.
// Output written f16 in sequence layout [s][h*64+d] for the final GEMM.
// ---------------------------------------------------------------------------
__global__ __launch_bounds__(256)
void sta3d_attn_kernel(const _Float16* __restrict__ Q, const _Float16* __restrict__ K,
                       const _Float16* __restrict__ VT, _Float16* __restrict__ O) {
  __shared__ __align__(16) _Float16 ldsP[8][16 * 136];  // per-wave P slab

  const int tid = threadIdx.x, wave = tid >> 5, lane = tid & 31;
  const int h  = blockIdx.x >> 6;
  const int nt = blockIdx.x & 63;
  const int ntt = nt >> 4, nth = (nt >> 2) & 3, ntw = nt & 3;

  // window bases: T dim (nt=4,w=3): clamp(i,1,2)-1 ; H/W dims (nt=4,w=2): clamp(i,1,3)-1
  int ct = ntt; if (ct < 1) ct = 1; if (ct > 2) ct = 2; ct -= 1;
  int ch = nth; if (ch < 1) ch = 1; if (ch > 3) ch = 3; ch -= 1;
  int cw = ntw; if (cw < 1) cw = 1; if (cw > 3) cw = 3; cw -= 1;

  const _Float16* qbase = Q + (size_t)(h * 64 + nt) * 128 * 64;
  const int qrow = wave * 16;

  Frag aq[2];
  aq[0] = load_frag(qbase, 64, qrow, 0);
  aq[1] = load_frag(qbase, 64, qrow, 32);

  v8f o_acc[4];
#pragma unroll
  for (int dc = 0; dc < 4; ++dc) o_acc[dc] = (v8f){};
  float m[8], l[8];
#pragma unroll
  for (int r = 0; r < 8; ++r) { m[r] = -__builtin_inff(); l[r] = 0.0f; }

  _Float16* myP = &ldsP[wave][0];

  for (int a = 0; a < 3; ++a) {
    for (int b = 0; b < 2; ++b) {
      for (int c = 0; c < 2; ++c) {
        const int kt = ((ct + a) * 4 + (ch + b)) * 4 + (cw + c);
        const _Float16* kbase = K  + (size_t)(h * 64 + kt) * 128 * 64;
        const _Float16* vbase = VT + (size_t)(h * 64 + kt) * 64 * 128;

        // scores: 16 (q rows) x 128 (keys), fp32 accum
        v8f s[8];
#pragma unroll
        for (int nc = 0; nc < 8; ++nc) s[nc] = (v8f){};
#pragma unroll
        for (int nc = 0; nc < 8; ++nc) {
          Frag bk0 = load_frag(kbase, 64, nc * 16, 0);
          Frag bk1 = load_frag(kbase, 64, nc * 16, 32);
          s[nc] = wmma16(aq[0], bk0, s[nc]);
          s[nc] = wmma16(aq[1], bk1, s[nc]);
        }

        // streaming softmax update (rows M=r / M=r+8 live in lane halves)
#pragma unroll
        for (int r = 0; r < 8; ++r) {
          float mx = -__builtin_inff();
#pragma unroll
          for (int nc = 0; nc < 8; ++nc) {
            s[nc][r] *= 0.125f;  // 1/sqrt(64)
            mx = fmaxf(mx, s[nc][r]);
          }
#pragma unroll
          for (int off = 1; off < 16; off <<= 1) mx = fmaxf(mx, __shfl_xor(mx, off, 32));
          const float mnew = fmaxf(m[r], mx);
          const float corr = __expf(m[r] - mnew);
          float rs = 0.0f;
#pragma unroll
          for (int nc = 0; nc < 8; ++nc) {
            const float p = __expf(s[nc][r] - mnew);
            s[nc][r] = p;
            rs += p;
          }
#pragma unroll
          for (int off = 1; off < 16; off <<= 1) rs += __shfl_xor(rs, off, 32);
          l[r] = l[r] * corr + rs;
          m[r] = mnew;
#pragma unroll
          for (int dc = 0; dc < 4; ++dc) o_acc[dc][r] *= corr;
        }

        // transpose P (C layout -> A layout) through the per-wave LDS slab
        {
          const int mo   = (lane & 16) ? 8 : 0;
          const int ncol = lane & 15;
#pragma unroll
          for (int nc = 0; nc < 8; ++nc)
#pragma unroll
            for (int r = 0; r < 8; ++r)
              myP[(mo + r) * 136 + nc * 16 + ncol] = (_Float16)s[nc][r];
        }
        // same-wave DS ops are in-order; dep waits handled by DScnt

        Frag ap[4];
#pragma unroll
        for (int kc = 0; kc < 4; ++kc) ap[kc] = load_frag(myP, 136, 0, kc * 32);
#pragma unroll
        for (int dc = 0; dc < 4; ++dc) {
#pragma unroll
          for (int kc = 0; kc < 4; ++kc) {
            Frag bv = load_frag(vbase, 128, dc * 16, kc * 32);
            o_acc[dc] = wmma16(ap[kc], bv, o_acc[dc]);
          }
        }
      }
    }
  }

  // finalize: o /= l ; write f16 in sequence layout for the final projection
  const int mo   = (lane & 16) ? 8 : 0;
  const int dcol = lane & 15;
#pragma unroll
  for (int dc = 0; dc < 4; ++dc) {
#pragma unroll
    for (int r = 0; r < 8; ++r) {
      const int p  = qrow + mo + r;
      const int tt = p >> 6, th = (p >> 3) & 7, tw = p & 7;
      const int srow = ntt * 2048 + tt * 1024 + nth * 256 + th * 32 + ntw * 8 + tw;
      O[(size_t)srow * 1024 + h * 64 + dc * 16 + dcol] =
          (_Float16)(o_acc[dc][r] / l[r]);
    }
  }
}

// ---------------------------------------------------------------------------
extern "C" void kernel_launch(void* const* d_in, const int* in_sizes, int n_in,
                              void* d_out, int out_size, void* d_ws, size_t ws_size,
                              hipStream_t stream) {
  (void)in_sizes; (void)n_in; (void)out_size; (void)ws_size;
  const float* hs = (const float*)d_in[0];
  const float* wq = (const float*)d_in[1];
  const float* wk = (const float*)d_in[2];
  const float* wv = (const float*)d_in[3];
  const float* wo = (const float*)d_in[4];
  float* out = (float*)d_out;

  char* ws = (char*)d_ws;
  const size_t MB = (size_t)1024 * 1024;
  _Float16* HS16 = (_Float16*)(ws);             // 16 MB: [8192][1024] f16
  _Float16* WQT  = (_Float16*)(ws + 16 * MB);   // 2 MB each: [n][k] f16
  _Float16* WKT  = (_Float16*)(ws + 18 * MB);
  _Float16* WVT  = (_Float16*)(ws + 20 * MB);
  _Float16* WOT  = (_Float16*)(ws + 22 * MB);
  _Float16* Q16  = (_Float16*)(ws + 24 * MB);   // 16 MB each
  _Float16* K16  = (_Float16*)(ws + 40 * MB);
  _Float16* VT16 = (_Float16*)(ws + 56 * MB);
  _Float16* O16  = (_Float16*)(ws + 72 * MB);

  // prep: convert activations; convert+transpose weights (bandwidth-trivial)
  sta3d_cvt_kernel<<<4096, 256, 0, stream>>>(hs, HS16);
  sta3d_cvtT_kernel<<<256, 256, 0, stream>>>(wq, WQT);
  sta3d_cvtT_kernel<<<256, 256, 0, stream>>>(wk, WKT);
  sta3d_cvtT_kernel<<<256, 256, 0, stream>>>(wv, WVT);
  sta3d_cvtT_kernel<<<256, 256, 0, stream>>>(wo, WOT);

  // QKV projections (LDS-free WMMA GEMM, scattered into attention layout)
  sta3d_gemm_kernel<0><<<256, 256, 0, stream>>>(HS16, WQT, Q16, nullptr);
  sta3d_gemm_kernel<0><<<256, 256, 0, stream>>>(HS16, WKT, K16, nullptr);
  sta3d_gemm_kernel<1><<<256, 256, 0, stream>>>(HS16, WVT, VT16, nullptr);
  // windowed attention: one block per (head, q-tile)
  sta3d_attn_kernel<<<16 * 64, 256, 0, stream>>>(Q16, K16, VT16, O16);
  // output projection -> fp32
  sta3d_gemm_kernel<2><<<256, 256, 0, stream>>>(O16, WOT, nullptr, out);
}
